// TFMistralAttention_15461882266166
// MI455X (gfx1250) — matrix operand order repaired and checked by reading, not compiled, not run on hardware
//
#include <hip/hip_runtime.h>

typedef __attribute__((ext_vector_type(16))) _Float16 v16h;
typedef __attribute__((ext_vector_type(8)))  _Float16 v8h;
typedef __attribute__((ext_vector_type(4)))  _Float16 v4h;
typedef __attribute__((ext_vector_type(8)))  float    v8f;
typedef __attribute__((ext_vector_type(4)))  float    v4f;
typedef __attribute__((ext_vector_type(4)))  int      v4i;

#define S_LEN 2048
#define HID   4096
#define NH    32
#define NKV   8
#define HD    128
#define WIN   1024

// ---------------------------------------------------------------------------
// Async global->LDS staging (gfx1250 GLOBAL_LOAD_ASYNC_TO_LDS_B128, ASYNCcnt).
// Builtin signature (from compiler diagnostic): (v4i AS1*, v4i AS3*, imm, imm).
// Guarded fallback: plain 16B load + ds_store (DScnt, same barrier scheme).
// ---------------------------------------------------------------------------
#if __has_builtin(__builtin_amdgcn_global_load_async_to_lds_b128)
#define HAS_ASYNC_LDS 1
#else
#define HAS_ASYNC_LDS 0
#endif

static __device__ __forceinline__ void cp16_g2l(const _Float16* g, _Float16* l) {
#if HAS_ASYNC_LDS
  __builtin_amdgcn_global_load_async_to_lds_b128(
      (__attribute__((address_space(1))) v4i*)g,
      (__attribute__((address_space(3))) v4i*)l, 0, 0);
#else
  *(v8h*)l = *(const v8h*)g;
#endif
}

static __device__ __forceinline__ void async_stage_wait() {
#if HAS_ASYNC_LDS
#if __has_builtin(__builtin_amdgcn_s_wait_asynccnt)
  __builtin_amdgcn_s_wait_asynccnt(0);
#else
  asm volatile("s_wait_asynccnt 0x0" ::: "memory");
#endif
#endif
}

// ---------------------------------------------------------------------------
// WMMA fragment helpers (gfx1250, wave32, v_wmma_f32_16x16x32_f16)
// A 16x32 f16: lane r=lane&15 holds row r; elems 0..7 = K kh*8..+7,
//              elems 8..15 = K 16+kh*8..+7   (kh = lane>>4)
// B 32x16 f16: lane r holds column r; elems 0..7 = K kh*8..+7,
//              elems 8..15 = K 16+kh*8..+7   (same as A)
// C/D 16x16 f32: lane = N + 16*(M/8), vgpr g = M%8
// ---------------------------------------------------------------------------
static __device__ __forceinline__ v16h load_a_frag(const _Float16* row, int kh) {
  const v8h a = *(const v8h*)(row + kh * 8);
  const v8h b = *(const v8h*)(row + kh * 8 + 16);
  v16h r;
#pragma unroll
  for (int i = 0; i < 8; ++i) { r[i] = a[i]; r[i + 8] = b[i]; }
  return r;
}

static __device__ __forceinline__ v16h load_b_frag(const _Float16* row, int kh) {
  const v8h a = *(const v8h*)(row + kh * 8);
  const v8h b = *(const v8h*)(row + kh * 8 + 16);
  v16h r;
#pragma unroll
  for (int i = 0; i < 8; ++i) { r[i] = a[i]; r[i + 8] = b[i]; }
  return r;
}

static __device__ __forceinline__ v8f wmma16(v16h a, v16h b, v8f c) {
  return __builtin_amdgcn_wmma_f32_16x16x32_f16(false, a, false, b, (short)0, c,
                                                false, false);
}

// ---------------------------------------------------------------------------
// fp32 -> fp16 elementwise convert (n multiple of 4)
// ---------------------------------------------------------------------------
__global__ void cvt_f32_f16_kernel(const float* __restrict__ in,
                                   _Float16* __restrict__ out, int n) {
  int i = (blockIdx.x * blockDim.x + threadIdx.x) * 4;
  if (i + 3 < n) {
    v4f x = *(const v4f*)(in + i);
    v4h y;
#pragma unroll
    for (int e = 0; e < 4; ++e) y[e] = (_Float16)x[e];
    *(v4h*)(out + i) = y;
  }
}

// ---------------------------------------------------------------------------
// Convert + transpose: W[K][N] fp32 -> Wt[N][K] fp16 (32x32 LDS tiles)
// ---------------------------------------------------------------------------
__global__ __launch_bounds__(256) void transp_cvt_kernel(
    const float* __restrict__ W, _Float16* __restrict__ Wt, int K, int N) {
  __shared__ float tile[32][33];
  const int tx = threadIdx.x, ty = threadIdx.y;  // blockDim (32,8)
  const int n0 = blockIdx.x * 32, k0 = blockIdx.y * 32;
#pragma unroll
  for (int i = 0; i < 4; ++i)
    tile[ty + i * 8][tx] = W[(size_t)(k0 + ty + i * 8) * N + (n0 + tx)];
  __syncthreads();
#pragma unroll
  for (int i = 0; i < 4; ++i)
    Wt[(size_t)(n0 + ty + i * 8) * K + (k0 + tx)] =
        (_Float16)tile[tx][ty + i * 8];
}

// ---------------------------------------------------------------------------
// WMMA GEMM: C = A[M][K] * Bt[N][K]^T, f16 in / f32 accum.
// Block tile 128x128, BK=32, 8 waves (wave tile 32x64), double-buffered LDS,
// one barrier per K-step, async global->LDS staging.
// OUTMODE: 0 = f16 row-major, 1 = f32 row-major, 2 = f16 transposed [N][M]
// ---------------------------------------------------------------------------
#define KS 40  // 32 + 8 pad f16 (80B rows, 16B aligned)

template <int OUTMODE>
__global__ __launch_bounds__(256) void gemm_wmma_kernel(
    const _Float16* __restrict__ A, const _Float16* __restrict__ Bt,
    void* __restrict__ C, int M, int N, int K) {
  __shared__ _Float16 sA[2][128 * KS];
  __shared__ _Float16 sB[2][128 * KS];
  const int tid = threadIdx.x;
  const int lane = tid & 31, wave = tid >> 5;
  const int r = lane & 15, kh = lane >> 4;
  const int wm = (wave & 3) * 32, wn = (wave >> 2) * 64;
  const int rowBase = blockIdx.y * 128, colBase = blockIdx.x * 128;
  const int lr = tid >> 2;       // 0..63: tile row per pass
  const int lc = (tid & 3) * 8;  // 0,8,16,24

  auto stage = [&](int kk, int buf) {
#pragma unroll
    for (int p = 0; p < 2; ++p) {
      const int row = p * 64 + lr;
      cp16_g2l(&A[(size_t)(rowBase + row) * K + kk + lc],
               &sA[buf][row * KS + lc]);
      cp16_g2l(&Bt[(size_t)(colBase + row) * K + kk + lc],
               &sB[buf][row * KS + lc]);
    }
  };

  v8f acc[2][4];
#pragma unroll
  for (int mt = 0; mt < 2; ++mt)
#pragma unroll
    for (int nt = 0; nt < 4; ++nt)
#pragma unroll
      for (int g = 0; g < 8; ++g) acc[mt][nt][g] = 0.0f;

  stage(0, 0);
  const int nsteps = K / 32;
  for (int c = 0; c < nsteps; ++c) {
    async_stage_wait();
    __syncthreads();
    if (c + 1 < nsteps) stage((c + 1) * 32, (c + 1) & 1);

    const _Float16* bufA = sA[c & 1];
    const _Float16* bufB = sB[c & 1];
    v16h af[2], bf[4];
#pragma unroll
    for (int mt = 0; mt < 2; ++mt)
      af[mt] = load_a_frag(&bufA[(wm + mt * 16 + r) * KS], kh);
#pragma unroll
    for (int nt = 0; nt < 4; ++nt)
      bf[nt] = load_b_frag(&bufB[(wn + nt * 16 + r) * KS], kh);
#pragma unroll
    for (int mt = 0; mt < 2; ++mt)
#pragma unroll
      for (int nt = 0; nt < 4; ++nt)
        acc[mt][nt] = wmma16(af[mt], bf[nt], acc[mt][nt]);
  }

#pragma unroll
  for (int mt = 0; mt < 2; ++mt)
#pragma unroll
    for (int nt = 0; nt < 4; ++nt)
#pragma unroll
      for (int g = 0; g < 8; ++g) {
        const int row = rowBase + wm + mt * 16 + kh * 8 + g;
        const int col = colBase + wn + nt * 16 + r;
        if (OUTMODE == 1)
          ((float*)C)[(size_t)row * N + col] = acc[mt][nt][g];
        else if (OUTMODE == 0)
          ((_Float16*)C)[(size_t)row * N + col] = (_Float16)acc[mt][nt][g];
        else  // transposed f16: out[col][row], leading dim M
          ((_Float16*)C)[(size_t)col * M + row] = (_Float16)acc[mt][nt][g];
      }
}

// ---------------------------------------------------------------------------
// In-place RoPE on f16 [S][H*128]; position_ids == arange(S)
// ---------------------------------------------------------------------------
__global__ void rope_kernel(_Float16* __restrict__ X, int H) {
  const int id = blockIdx.x * blockDim.x + threadIdx.x;
  if (id >= S_LEN * H * 64) return;
  const int d = id & 63;
  const int h = (id >> 6) % H;
  const int s = id / (64 * H);
  const float inv = __powf(10000.0f, -(float)(2 * d) * (1.0f / 128.0f));
  float sn, cs;
  __sincosf((float)s * inv, &sn, &cs);
  const size_t base = (size_t)s * (H * 128) + h * 128 + d;
  const float x0 = (float)X[base], x1 = (float)X[base + 64];
  X[base]      = (_Float16)(x0 * cs - x1 * sn);
  X[base + 64] = (_Float16)(x1 * cs + x0 * sn);
}

// ---------------------------------------------------------------------------
// Flash attention w/ sliding window + GQA.
// Grid: (S/128 q-tiles, 32 heads). 256 threads = 8 waves, 16 q-rows/wave.
// V is pre-transposed globally (Vt[d_global][s]) so both K and V stage into
// LDS with 16B vector copies; KV chunks double-buffered, 1 barrier per chunk.
// ---------------------------------------------------------------------------
__global__ __launch_bounds__(256) void attn_kernel(
    const _Float16* __restrict__ Q, const _Float16* __restrict__ K,
    const _Float16* __restrict__ Vt, _Float16* __restrict__ O) {
  __shared__ _Float16 sK[2][32 * 136];   // 32 keys x 128 d (row-major, pad 8)
  __shared__ _Float16 sV[2][128 * 40];   // V^T: 128 d x 32 keys (pad 8)
  __shared__ _Float16 sP[8][16 * 40];    // per-wave P 16x32 (C -> A layout)

  const int h = blockIdx.y, kvh = h >> 2;
  const int q0 = blockIdx.x * 128;
  const int tid = threadIdx.x, lane = tid & 31, wave = tid >> 5;
  const int r = lane & 15, kh = lane >> 4;
  const float scale = 0.08838834764831845f;  // 1/sqrt(128)

  const int kr = tid >> 4, kc = (tid & 15) * 8;  // K staging: 16 rows/pass
  const int vr = tid >> 2, vc = (tid & 3) * 8;   // Vt staging: 64 rows/pass

  auto stage = [&](int j0, int buf) {
#pragma unroll
    for (int p = 0; p < 2; ++p) {
      const int key = p * 16 + kr;
      cp16_g2l(&K[(size_t)(j0 + key) * (NKV * HD) + kvh * HD + kc],
               &sK[buf][key * 136 + kc]);
      const int d = p * 64 + vr;
      cp16_g2l(&Vt[(size_t)(kvh * HD + d) * S_LEN + j0 + vc],
               &sV[buf][d * 40 + vc]);
    }
  };

  // Q A-fragments for this wave's 16 rows (head_dim in 4 chunks of 32)
  v16h qf[4];
  {
    const _Float16* qrow = Q + (size_t)(q0 + wave * 16 + r) * HID + h * HD;
#pragma unroll
    for (int f = 0; f < 4; ++f) qf[f] = load_a_frag(qrow + f * 32, kh);
  }

  v8f oacc[8];
  float mrow[8], lrow[8];
#pragma unroll
  for (int dt = 0; dt < 8; ++dt)
#pragma unroll
    for (int g = 0; g < 8; ++g) oacc[dt][g] = 0.0f;
#pragma unroll
  for (int g = 0; g < 8; ++g) { mrow[g] = -1e30f; lrow[g] = 0.0f; }

  int jlo = q0 - (WIN - 1);
  if (jlo < 0) jlo = 0;
  jlo &= ~31;
  const int jhi = q0 + 128;

  stage(jlo, 0);
  int c = 0;
  for (int j0 = jlo; j0 < jhi; j0 += 32, ++c) {
    async_stage_wait();
    __syncthreads();
    if (j0 + 32 < jhi) stage(j0 + 32, (c + 1) & 1);
    const _Float16* bK = sK[c & 1];
    const _Float16* bV = sV[c & 1];

    // scores: two 16x16 key tiles, K-dim = head_dim in 4 steps of 32
    v8f sc[2];
#pragma unroll
    for (int t = 0; t < 2; ++t) {
#pragma unroll
      for (int g = 0; g < 8; ++g) sc[t][g] = 0.0f;
#pragma unroll
      for (int f = 0; f < 4; ++f)
        sc[t] = wmma16(qf[f], load_b_frag(&bK[(t * 16 + r) * 136 + f * 32], kh),
                       sc[t]);
    }

    // online softmax (row stats; a q-row spans the 16 lanes of one half)
    float alpha[8];
    const int ja = j0 + r, jb = j0 + 16 + r;
#pragma unroll
    for (int g = 0; g < 8; ++g) {
      const int i = q0 + wave * 16 + kh * 8 + g;
      const bool aa = (ja <= i) && (ja > i - WIN);
      const bool ab = (jb <= i) && (jb > i - WIN);
      const float sa = sc[0][g] * scale, sb = sc[1][g] * scale;
      float cm = fmaxf(aa ? sa : -1e30f, ab ? sb : -1e30f);
#pragma unroll
      for (int off = 1; off < 16; off <<= 1)
        cm = fmaxf(cm, __shfl_xor(cm, off, 32));
      const float mo = mrow[g];
      const float mn = fmaxf(mo, cm);
      const float al = __expf(mo - mn);
      const float pa = aa ? __expf(sa - mn) : 0.0f;
      const float pb = ab ? __expf(sb - mn) : 0.0f;
      float ps = pa + pb;
#pragma unroll
      for (int off = 1; off < 16; off <<= 1) ps += __shfl_xor(ps, off, 32);
      mrow[g] = mn;
      lrow[g] = lrow[g] * al + ps;
      alpha[g] = al;
      sP[wave][(kh * 8 + g) * 40 + r]      = (_Float16)pa;
      sP[wave][(kh * 8 + g) * 40 + 16 + r] = (_Float16)pb;
    }

#pragma unroll
    for (int dt = 0; dt < 8; ++dt)
#pragma unroll
      for (int g = 0; g < 8; ++g) oacc[dt][g] *= alpha[g];

    // wave-internal LDS ordering: P stores visible before A-fragment read
    asm volatile("s_wait_dscnt 0x0" ::: "memory");
    const v16h pf = load_a_frag(&sP[wave][r * 40], kh);
#pragma unroll
    for (int dt = 0; dt < 8; ++dt)
      oacc[dt] = wmma16(pf, load_b_frag(&bV[(dt * 16 + r) * 40], kh), oacc[dt]);
  }

#pragma unroll
  for (int dt = 0; dt < 8; ++dt)
#pragma unroll
    for (int g = 0; g < 8; ++g) {
      const int i = q0 + wave * 16 + kh * 8 + g;
      const int col = h * HD + dt * 16 + r;
      O[(size_t)i * HID + col] = (_Float16)(oacc[dt][g] / lrow[g]);
    }
}

// ---------------------------------------------------------------------------
extern "C" void kernel_launch(void* const* d_in, const int* in_sizes, int n_in,
                              void* d_out, int out_size, void* d_ws,
                              size_t ws_size, hipStream_t stream) {
  (void)in_sizes; (void)n_in; (void)out_size; (void)ws_size;
  const float* hidden = (const float*)d_in[0];
  // d_in[1] attention_mask, d_in[2] position_ids: recomputed analytically
  const float* Wq = (const float*)d_in[3];
  const float* Wk = (const float*)d_in[4];
  const float* Wv = (const float*)d_in[5];
  const float* Wo = (const float*)d_in[6];
  float* out = (float*)d_out;

  _Float16* p = (_Float16*)d_ws;
  _Float16* Xh  = p; p += (size_t)S_LEN * HID;
  _Float16* WqT = p; p += (size_t)HID * HID;
  _Float16* WkT = p; p += (size_t)(NKV * HD) * HID;
  _Float16* WvT = p; p += (size_t)(NKV * HD) * HID;
  _Float16* WoT = p; p += (size_t)HID * HID;
  _Float16* Qb  = p; p += (size_t)S_LEN * HID;
  _Float16* Kb  = p; p += (size_t)S_LEN * (NKV * HD);
  _Float16* VbT = p; p += (size_t)(NKV * HD) * S_LEN;  // transposed [d][s]
  _Float16* Ab  = p; p += (size_t)S_LEN * HID;

  {
    const int n = S_LEN * HID;
    cvt_f32_f16_kernel<<<n / 1024, 256, 0, stream>>>(hidden, Xh, n);
  }
  transp_cvt_kernel<<<dim3(HID / 32, HID / 32), dim3(32, 8), 0, stream>>>(
      Wq, WqT, HID, HID);
  transp_cvt_kernel<<<dim3((NKV * HD) / 32, HID / 32), dim3(32, 8), 0, stream>>>(
      Wk, WkT, HID, NKV * HD);
  transp_cvt_kernel<<<dim3((NKV * HD) / 32, HID / 32), dim3(32, 8), 0, stream>>>(
      Wv, WvT, HID, NKV * HD);
  transp_cvt_kernel<<<dim3(HID / 32, HID / 32), dim3(32, 8), 0, stream>>>(
      Wo, WoT, HID, HID);

  gemm_wmma_kernel<0><<<dim3(HID / 128, S_LEN / 128), 256, 0, stream>>>(
      Xh, WqT, Qb, S_LEN, HID, HID);
  gemm_wmma_kernel<0><<<dim3((NKV * HD) / 128, S_LEN / 128), 256, 0, stream>>>(
      Xh, WkT, Kb, S_LEN, NKV * HD, HID);
  gemm_wmma_kernel<2><<<dim3((NKV * HD) / 128, S_LEN / 128), 256, 0, stream>>>(
      Xh, WvT, VbT, S_LEN, NKV * HD, HID);  // V written transposed

  rope_kernel<<<(S_LEN * NH * 64) / 256, 256, 0, stream>>>(Qb, NH);
  rope_kernel<<<(S_LEN * NKV * 64) / 256, 256, 0, stream>>>(Kb, NKV);

  attn_kernel<<<dim3(S_LEN / 128, NH), 256, 0, stream>>>(Qb, Kb, VbT, Ab);

  gemm_wmma_kernel<1><<<dim3(HID / 128, S_LEN / 128), 256, 0, stream>>>(
      Ab, WoT, out, S_LEN, HID, HID);
}
